// ConvViT_62448824484517
// MI455X (gfx1250) — matrix operand order repaired and checked by reading, not compiled, not run on hardware
//
#include <hip/hip_runtime.h>
#include <hip/hip_bf16.h>

typedef __bf16 bf16_t;
typedef __attribute__((ext_vector_type(16))) __bf16 v16bf;
typedef __attribute__((ext_vector_type(8)))  __bf16 v8bf;
typedef __attribute__((ext_vector_type(8)))  float  v8f;

static __device__ __forceinline__ v16bf join8(v8bf lo, v8bf hi) {
  return __builtin_shufflevector(lo, hi, 0,1,2,3,4,5,6,7,8,9,10,11,12,13,14,15);
}
static __device__ __forceinline__ v8f wmma_bf16(v16bf a, v16bf b, v8f c) {
  return __builtin_amdgcn_wmma_f32_16x16x32_bf16(false, a, false, b, (short)0, c, false, false);
}
static __device__ __forceinline__ v8f vzero8() {
  v8f z;
#pragma unroll
  for (int i = 0; i < 8; ++i) z[i] = 0.0f;
  return z;
}

// ---------------- problem constants ----------------
// B=8, H=W=64, D=768, HEADS=12, HD=64, WS=14, N=196, MLP=3072
// windows: 5x5 per image -> 200 windows; padded token dims: 208 (q/k), 224 (v)

// ---------------- fp32 -> bf16 convert ----------------
__global__ void cvt_bf16_kernel(const float* __restrict__ s, bf16_t* __restrict__ d, int n) {
  int i = blockIdx.x * 256 + threadIdx.x;
  if (i < n) d[i] = (bf16_t)s[i];
}

// ---------------- LayerNorm (row = 768) ----------------
// MODE 0: window-partition layout with zero padding, rows = 200*196
// MODE 1: plain rows = 32768
template <int MODE>
__global__ __launch_bounds__(256) void ln_kernel(const float* __restrict__ x,
                                                 const float* __restrict__ g,
                                                 const float* __restrict__ b,
                                                 bf16_t* __restrict__ dst) {
  int row = blockIdx.x;
  int tid = threadIdx.x;
  size_t srcoff, dstoff;
  bool pad = false;
  if (MODE == 0) {
    int win = row / 196, n = row % 196;
    int bb = win / 25, wh = (win % 25) / 5, ww = win % 5;
    int i = n / 14, j = n % 14;
    int h = wh * 14 + i, w = ww * 14 + j;
    pad = (h >= 64) || (w >= 64);
    srcoff = (((size_t)bb * 64 + h) * 64 + w) * 768;
    dstoff = (size_t)row * 768;
  } else {
    srcoff = (size_t)row * 768;
    dstoff = srcoff;
  }
  bf16_t* out = dst + dstoff;
  if (MODE == 0 && pad) {
    for (int c = tid; c < 768; c += 256) out[c] = (bf16_t)0.0f;
    return;
  }
  const float* src = x + srcoff;
  float v0 = src[tid], v1 = src[tid + 256], v2 = src[tid + 512];
  float s = v0 + v1 + v2;
  float q = v0 * v0 + v1 * v1 + v2 * v2;
#pragma unroll
  for (int o = 16; o > 0; o >>= 1) {
    s += __shfl_xor(s, o, 32);
    q += __shfl_xor(q, o, 32);
  }
  __shared__ float rs[8], rq[8], mb[2];
  int wid = tid >> 5, ln = tid & 31;
  if (ln == 0) { rs[wid] = s; rq[wid] = q; }
  __syncthreads();
  if (tid == 0) {
    float S = 0.f, Q = 0.f;
    for (int k = 0; k < 8; ++k) { S += rs[k]; Q += rq[k]; }
    float mu = S * (1.0f / 768.0f);
    float var = Q * (1.0f / 768.0f) - mu * mu;
    mb[0] = mu;
    mb[1] = rsqrtf(var + 1e-5f);
  }
  __syncthreads();
  float mu = mb[0], r = mb[1];
  out[tid]       = (bf16_t)((v0 - mu) * r * g[tid]       + b[tid]);
  out[tid + 256] = (bf16_t)((v1 - mu) * r * g[tid + 256] + b[tid + 256]);
  out[tid + 512] = (bf16_t)((v2 - mu) * r * g[tid + 512] + b[tid + 512]);
}

// ---------------- WMMA GEMM: C[M,N] = A[M,K] @ W[K,N] (+epilogue) ----------------
// block tile 128x128, 8 waves (4 M x 2 N), each wave 32x64, K-step 32.
// MODE 0: QKV scatter (+bias) into qT/k/v attention layouts
// MODE 1: proj (+bias) + residual + window-unpartition crop -> fp32 out
// MODE 2: fc1 (+bias) + exact GELU -> bf16
// MODE 3: fc2 (+bias) + residual (in-place on o_f32) -> fp32
template <int MODE>
__global__ __launch_bounds__(256) void gemm_kernel(
    const bf16_t* __restrict__ A, const bf16_t* __restrict__ W,
    int M, int K, int N, const float* __restrict__ bias,
    bf16_t* __restrict__ o_qT, bf16_t* __restrict__ o_k, bf16_t* __restrict__ o_v,
    const float* __restrict__ resid, float* __restrict__ o_f32,
    bf16_t* __restrict__ o_bf) {
  __shared__ __align__(16) bf16_t a_s[128 * 40];
  __shared__ __align__(16) bf16_t b_s[32 * 136];
  int tid = threadIdx.x;
  int ln = tid & 31, wid = tid >> 5;
  int lo16 = ln & 15, hi = ln >> 4;
  int n0 = blockIdx.x * 128, m0 = blockIdx.y * 128;
  int wm = wid & 3, wn = wid >> 2;

  v8f acc[2][4];
#pragma unroll
  for (int i = 0; i < 2; ++i)
#pragma unroll
    for (int j = 0; j < 4; ++j) acc[i][j] = vzero8();

  int arow = tid >> 1, achk = tid & 1;  // A: 128 rows x 2 chunks of 32B
  int brow = tid >> 3, bchk = tid & 7;  // B: 32 rows x 8 chunks of 32B

  for (int k0 = 0; k0 < K; k0 += 32) {
    {
      int gr = m0 + arow;
      uint4 z0 = make_uint4(0, 0, 0, 0), z1 = z0;
      if (gr < M) {
        const uint4* p = (const uint4*)(A + (size_t)gr * K + k0 + achk * 16);
        z0 = p[0];
        z1 = p[1];
        if (k0 + 32 < K) __builtin_prefetch((const void*)(A + (size_t)gr * K + k0 + 32), 0, 1);
      }
      uint4* d = (uint4*)(a_s + arow * 40 + achk * 16);
      d[0] = z0; d[1] = z1;
    }
    {
      const uint4* p = (const uint4*)(W + (size_t)(k0 + brow) * N + n0 + bchk * 16);
      uint4* d = (uint4*)(b_s + brow * 136 + bchk * 16);
      d[0] = p[0]; d[1] = p[1];
    }
    __syncthreads();
    v16bf af[2], bfrag[4];
#pragma unroll
    for (int mt = 0; mt < 2; ++mt) {
      const bf16_t* p = a_s + (wm * 32 + mt * 16 + lo16) * 40 + hi * 8;
      af[mt] = join8(*(const v8bf*)p, *(const v8bf*)(p + 16));
    }
#pragma unroll
    for (int nt = 0; nt < 4; ++nt) {
      const bf16_t* p = b_s + wn * 64 + nt * 16 + lo16;
#pragma unroll
      for (int e = 0; e < 16; ++e) bfrag[nt][e] = p[((e & 7) + hi * 8 + (e >> 3) * 16) * 136];
    }
#pragma unroll
    for (int mt = 0; mt < 2; ++mt)
#pragma unroll
      for (int nt = 0; nt < 4; ++nt)
        acc[mt][nt] = wmma_bf16(af[mt], bfrag[nt], acc[mt][nt]);
    __syncthreads();
  }

#pragma unroll
  for (int mt = 0; mt < 2; ++mt)
#pragma unroll
    for (int nt = 0; nt < 4; ++nt)
#pragma unroll
      for (int r = 0; r < 8; ++r) {
        int m = m0 + wm * 32 + mt * 16 + hi * 8 + r;
        int n = n0 + wn * 64 + nt * 16 + lo16;
        if (m >= M) continue;
        float val = acc[mt][nt][r] + bias[n];
        if (MODE == 0) {
          int which = n / 768;
          int hd = (n % 768) / 64;
          int c = n & 63;
          int win = m / 196, qi = m % 196;
          int whd = win * 12 + hd;
          bf16_t v = (bf16_t)val;
          if (which == 0) {
            o_qT[((size_t)whd * 64 + c) * 208 + qi] = v;
          } else if (which == 1) {
            o_k[((size_t)whd * 208 + qi) * 64 + c] = v;
          } else {
            o_v[((size_t)whd * 224 + qi) * 64 + c] = v;
          }
        } else if (MODE == 1) {
          int win = m / 196, qi = m % 196;
          int bb = win / 25, wh2 = (win % 25) / 5, ww = win % 5;
          int i = qi / 14, j = qi % 14;
          int h = wh2 * 14 + i, w = ww * 14 + j;
          if (h < 64 && w < 64) {
            size_t idx = (((size_t)bb * 64 + h) * 64 + w) * 768 + n;
            o_f32[idx] = resid[idx] + val;
          }
        } else if (MODE == 2) {
          float gv = 0.5f * val * (1.0f + erff(val * 0.70710678118f));
          o_bf[(size_t)m * N + n] = (bf16_t)gv;
        } else {
          size_t idx = (size_t)m * N + n;
          o_f32[idx] = o_f32[idx] + val;
        }
      }
}

// ---------------- rel-pos bias via WMMA ----------------
// One block per (window, head). rel[qi, m] = q[qi,:] . rel_pos[m,:], m = 0..26.
// D[M=m, N=qi] = rel_pos_rows (A, LDS) x q^T (B, global zero-padded layout).
// Store with lookup applied: rel_hb[whd][qi][kh] = rel[qi, qi/14 + 13 - kh].
__global__ __launch_bounds__(416) void rel_kernel(
    const bf16_t* __restrict__ qT_g,
    const float* __restrict__ rph, const float* __restrict__ rpw,
    float* __restrict__ relhb, float* __restrict__ relwb) {
  __shared__ __align__(16) bf16_t rh_s[32 * 72];
  __shared__ __align__(16) bf16_t rw_s[32 * 72];
  int whd = blockIdx.x;
  int tid = threadIdx.x;
  for (int idx = tid; idx < 32 * 64; idx += 416) {
    int row = idx >> 6, col = idx & 63;
    float vh = (row < 27) ? rph[row * 64 + col] : 0.0f;
    float vw = (row < 27) ? rpw[row * 64 + col] : 0.0f;
    rh_s[row * 72 + col] = (bf16_t)vh;
    rw_s[row * 72 + col] = (bf16_t)vw;
  }
  __syncthreads();
  int ln = tid & 31, wid = tid >> 5;
  int lo16 = ln & 15, hi = ln >> 4;
  int qi0 = wid * 16;

  v16bf bq[2];
  const bf16_t* qTb = qT_g + (size_t)whd * 64 * 208;
#pragma unroll
  for (int s = 0; s < 2; ++s) {
    const bf16_t* p = qTb + (size_t)(s * 32) * 208 + qi0 + lo16;
#pragma unroll
    for (int e = 0; e < 16; ++e) bq[s][e] = p[(size_t)((e & 7) + hi * 8 + (e >> 3) * 16) * 208];
  }

  v8f aH[2], aW[2];
#pragma unroll
  for (int t = 0; t < 2; ++t) {
    v8f ah = vzero8(), aw = vzero8();
#pragma unroll
    for (int s = 0; s < 2; ++s) {
      const bf16_t* ph = rh_s + (t * 16 + lo16) * 72 + s * 32 + hi * 8;
      v16bf fh = join8(*(const v8bf*)ph, *(const v8bf*)(ph + 16));
      ah = wmma_bf16(fh, bq[s], ah);
      const bf16_t* pw = rw_s + (t * 16 + lo16) * 72 + s * 32 + hi * 8;
      v16bf fw = join8(*(const v8bf*)pw, *(const v8bf*)(pw + 16));
      aw = wmma_bf16(fw, bq[s], aw);
    }
    aH[t] = ah;
    aW[t] = aw;
  }

  int qi = qi0 + lo16;
  int hh = qi / 14, ww = qi % 14;
  float* oh = relhb + ((size_t)whd * 208 + qi) * 16;
  float* ow = relwb + ((size_t)whd * 208 + qi) * 16;
#pragma unroll
  for (int t = 0; t < 2; ++t)
#pragma unroll
    for (int r = 0; r < 8; ++r) {
      int m = t * 16 + hi * 8 + r;
      int kh = hh + 13 - m;
      if (kh >= 0 && kh < 14) oh[kh] = aH[t][r];
      int kw = ww + 13 - m;
      if (kw >= 0 && kw < 14) ow[kw] = aW[t][r];
    }
}

// ---------------- attention: one block per (window, head) ----------------
// 13 waves, wave w owns query columns qi in [16w, 16w+16). Computes S^T = K @ Q^T
// so softmaxed P lands directly in WMMA A-fragment layout for P @ V.
__global__ __launch_bounds__(416) void attn_kernel(
    const bf16_t* __restrict__ qT_g,
    const bf16_t* __restrict__ k_g, const bf16_t* __restrict__ v_g,
    const float* __restrict__ relhb, const float* __restrict__ relwb,
    bf16_t* __restrict__ ao) {
  __shared__ __align__(16) bf16_t k_s[208 * 72];
  __shared__ __align__(16) bf16_t v_s[224 * 72];
  int whd = blockIdx.x;
  int head = whd % 12, win = whd / 12;
  int tid = threadIdx.x;

  const uint4* ksrc = (const uint4*)(k_g + (size_t)whd * 208 * 64);
  for (int idx = tid; idx < 208 * 8; idx += 416) {
    int row = idx >> 3, ch = idx & 7;
    *(uint4*)(k_s + row * 72 + ch * 8) = ksrc[idx];
  }
  const uint4* vsrc = (const uint4*)(v_g + (size_t)whd * 224 * 64);
  for (int idx = tid; idx < 224 * 8; idx += 416) {
    int row = idx >> 3, ch = idx & 7;
    *(uint4*)(v_s + row * 72 + ch * 8) = vsrc[idx];
  }
  __syncthreads();

  int ln = tid & 31, wid = tid >> 5;
  int lo16 = ln & 15, hi = ln >> 4;
  int qi0 = wid * 16;

  // Q^T B-fragments (from global, zero-padded layout): lane = qi column, 16 c in table order
  v16bf bq[2];
  const bf16_t* qTb = qT_g + (size_t)whd * 64 * 208;
#pragma unroll
  for (int s = 0; s < 2; ++s) {
    const bf16_t* p = qTb + (size_t)(s * 32) * 208 + qi0 + lo16;
#pragma unroll
    for (int e = 0; e < 16; ++e) bq[s][e] = p[(size_t)((e & 7) + hi * 8 + (e >> 3) * 16) * 208];
  }

  // precomputed rel-pos bias for this lane's query (qi = qi0 + lo16)
  float relh[14], relw[14];
  {
    int qi = qi0 + lo16;
    const float* oh = relhb + ((size_t)whd * 208 + qi) * 16;
    const float* ow = relwb + ((size_t)whd * 208 + qi) * 16;
#pragma unroll
    for (int kk = 0; kk < 14; ++kk) {
      relh[kk] = oh[kk];
      relw[kk] = ow[kk];
    }
  }

  // S^T tiles: D[M=kj, N=qi] = K_tile x Q^T
  v8f accS[13];
#pragma unroll
  for (int t = 0; t < 13; ++t) {
    v8f acc = vzero8();
#pragma unroll
    for (int s = 0; s < 2; ++s) {
      const bf16_t* p = k_s + (t * 16 + lo16) * 72 + s * 32 + hi * 8;
      v16bf ak = join8(*(const v8bf*)p, *(const v8bf*)(p + 16));
      acc = wmma_bf16(ak, bq[s], acc);
    }
    accS[t] = acc;
  }

  // scale + bias + mask, then softmax over kj (split across lane pairs ln / ln+16)
  float mx = -1e30f;
#pragma unroll
  for (int t = 0; t < 13; ++t)
#pragma unroll
    for (int r = 0; r < 8; ++r) {
      int kj = t * 16 + hi * 8 + r;
      float f;
      if (kj >= 196) {
        f = -1e30f;
      } else {
        int kh = kj / 14, kw = kj - kh * 14;
        f = accS[t][r] * 0.125f + relh[kh] + relw[kw];
      }
      accS[t][r] = f;
      mx = fmaxf(mx, f);
    }
  mx = fmaxf(mx, __shfl_xor(mx, 16, 32));
  float sum = 0.f;
#pragma unroll
  for (int t = 0; t < 13; ++t)
#pragma unroll
    for (int r = 0; r < 8; ++r) {
      float p = __expf(accS[t][r] - mx);
      accS[t][r] = p;
      sum += p;
    }
  sum += __shfl_xor(sum, 16, 32);
  float inv = 1.0f / sum;

  // out = P @ V ; P A-fragments come straight from accS (no cross-lane movement)
  v8f accO[4];
#pragma unroll
  for (int ct = 0; ct < 4; ++ct) accO[ct] = vzero8();
#pragma unroll
  for (int s = 0; s < 7; ++s) {
    v16bf ap;
#pragma unroll
    for (int i = 0; i < 8; ++i) {
      ap[i] = (bf16_t)accS[2 * s][i];
      ap[8 + i] = (2 * s + 1 < 13) ? (bf16_t)accS[2 * s + 1][i] : (bf16_t)0.0f;
    }
#pragma unroll
    for (int ct = 0; ct < 4; ++ct) {
      const bf16_t* p = v_s + (s * 32) * 72 + ct * 16 + lo16;
      v16bf bv;
#pragma unroll
      for (int e = 0; e < 16; ++e) bv[e] = p[((e & 7) + hi * 8 + (e >> 3) * 16) * 72];
      accO[ct] = wmma_bf16(ap, bv, accO[ct]);
    }
  }

#pragma unroll
  for (int r = 0; r < 8; ++r) {
    float invr = __shfl(inv, r + 8 * hi, 32);
    int qi = qi0 + hi * 8 + r;
    if (qi < 196) {
#pragma unroll
      for (int ct = 0; ct < 4; ++ct) {
        int c = ct * 16 + lo16;
        ao[((size_t)win * 196 + qi) * 768 + head * 64 + c] = (bf16_t)(accO[ct][r] * invr);
      }
    }
  }
}

// ---------------- workspace layout (bytes) ----------------
static constexpr size_t SZ_XW  = 39200ULL * 768 * 2;
static constexpr size_t SZ_QT  = 2400ULL * 64 * 208 * 2;
static constexpr size_t SZ_KG  = 2400ULL * 208 * 64 * 2;
static constexpr size_t SZ_VG  = 2400ULL * 224 * 64 * 2;
static constexpr size_t SZ_RH  = 2400ULL * 208 * 16 * 4;
static constexpr size_t SZ_RW  = 2400ULL * 208 * 16 * 4;
static constexpr size_t SZ_AO  = 39200ULL * 768 * 2;
static constexpr size_t SZ_XN2 = 32768ULL * 768 * 2;
static constexpr size_t SZ_H   = 32768ULL * 3072 * 2;
static constexpr size_t SZ_WQ  = 768ULL * 2304 * 2;
static constexpr size_t SZ_WP  = 768ULL * 768 * 2;
static constexpr size_t SZ_W1  = 768ULL * 3072 * 2;
static constexpr size_t SZ_W2  = 3072ULL * 768 * 2;

static constexpr size_t OFF_XW  = 0;
static constexpr size_t OFF_QT  = OFF_XW + SZ_XW;
static constexpr size_t OFF_KG  = OFF_QT + SZ_QT;
static constexpr size_t OFF_VG  = OFF_KG + SZ_KG;
static constexpr size_t OFF_RH  = OFF_VG + SZ_VG;
static constexpr size_t OFF_RW  = OFF_RH + SZ_RH;
static constexpr size_t OFF_AO  = OFF_RW + SZ_RW;
static constexpr size_t OFF_XN2 = OFF_AO + SZ_AO;
static constexpr size_t OFF_H   = OFF_XN2 + SZ_XN2;
static constexpr size_t OFF_WQ  = OFF_H + SZ_H;
static constexpr size_t OFF_WP  = OFF_WQ + SZ_WQ;
static constexpr size_t OFF_W1  = OFF_WP + SZ_WP;
static constexpr size_t OFF_W2  = OFF_W1 + SZ_W1;

extern "C" void kernel_launch(void* const* d_in, const int* in_sizes, int n_in,
                              void* d_out, int out_size, void* d_ws, size_t ws_size,
                              hipStream_t stream) {
  (void)in_sizes; (void)n_in; (void)out_size; (void)ws_size;
  const float* x      = (const float*)d_in[0];
  const float* qkv_w  = (const float*)d_in[1];
  const float* qkv_b  = (const float*)d_in[2];
  const float* proj_w = (const float*)d_in[3];
  const float* proj_b = (const float*)d_in[4];
  const float* rph    = (const float*)d_in[5];
  const float* rpw    = (const float*)d_in[6];
  const float* n1w    = (const float*)d_in[7];
  const float* n1b    = (const float*)d_in[8];
  const float* n2w    = (const float*)d_in[9];
  const float* n2b    = (const float*)d_in[10];
  const float* fc1_w  = (const float*)d_in[11];
  const float* fc1_b  = (const float*)d_in[12];
  const float* fc2_w  = (const float*)d_in[13];
  const float* fc2_b  = (const float*)d_in[14];
  float* out = (float*)d_out;

  char* ws = (char*)d_ws;
  bf16_t* xw   = (bf16_t*)(ws + OFF_XW);
  bf16_t* qTg  = (bf16_t*)(ws + OFF_QT);
  bf16_t* kg   = (bf16_t*)(ws + OFF_KG);
  bf16_t* vg   = (bf16_t*)(ws + OFF_VG);
  float*  relh = (float*)(ws + OFF_RH);
  float*  relw = (float*)(ws + OFF_RW);
  bf16_t* ao   = (bf16_t*)(ws + OFF_AO);
  bf16_t* xn2  = (bf16_t*)(ws + OFF_XN2);
  bf16_t* hb   = (bf16_t*)(ws + OFF_H);
  bf16_t* wq   = (bf16_t*)(ws + OFF_WQ);
  bf16_t* wp   = (bf16_t*)(ws + OFF_WP);
  bf16_t* w1   = (bf16_t*)(ws + OFF_W1);
  bf16_t* w2   = (bf16_t*)(ws + OFF_W2);

  // weight downconversion fp32 -> bf16
  cvt_bf16_kernel<<<(int)((768 * 2304 + 255) / 256), 256, 0, stream>>>(qkv_w, wq, 768 * 2304);
  cvt_bf16_kernel<<<(int)((768 * 768 + 255) / 256), 256, 0, stream>>>(proj_w, wp, 768 * 768);
  cvt_bf16_kernel<<<(int)((768 * 3072 + 255) / 256), 256, 0, stream>>>(fc1_w, w1, 768 * 3072);
  cvt_bf16_kernel<<<(int)((3072 * 768 + 255) / 256), 256, 0, stream>>>(fc2_w, w2, 3072 * 768);

  // zero-fill attention operand buffers (padding rows/cols must be 0)
  hipMemsetAsync(qTg, 0, SZ_QT, stream);
  hipMemsetAsync(kg, 0, SZ_KG, stream);
  hipMemsetAsync(vg, 0, SZ_VG, stream);

  // LN1 + window partition (with zero padding) -> bf16
  ln_kernel<0><<<39200, 256, 0, stream>>>(x, n1w, n1b, xw);

  // QKV GEMM -> scatter to attention layouts
  gemm_kernel<0><<<dim3(2304 / 128, 307), 256, 0, stream>>>(
      xw, wq, 39200, 768, 2304, qkv_b, qTg, kg, vg, nullptr, nullptr, nullptr);

  // decomposed rel-pos bias tables via WMMA
  rel_kernel<<<2400, 416, 0, stream>>>(qTg, rph, rpw, relh, relw);

  // attention per (window, head)
  attn_kernel<<<2400, 416, 0, stream>>>(qTg, kg, vg, relh, relw, ao);

  // proj + residual + unpartition crop -> x1 (fp32 in d_out)
  gemm_kernel<1><<<dim3(768 / 128, 307), 256, 0, stream>>>(
      ao, wp, 39200, 768, 768, proj_b, nullptr, nullptr, nullptr, x, out, nullptr);

  // LN2 -> bf16
  ln_kernel<1><<<32768, 256, 0, stream>>>(out, n2w, n2b, xn2);

  // fc1 + GELU -> bf16
  gemm_kernel<2><<<dim3(3072 / 128, 256), 256, 0, stream>>>(
      xn2, w1, 32768, 768, 3072, fc1_b, nullptr, nullptr, nullptr, nullptr, nullptr, hb);

  // fc2 + residual -> final fp32 (in-place add into d_out)
  gemm_kernel<3><<<dim3(768 / 128, 256), 256, 0, stream>>>(
      hb, w2, 32768, 3072, 768, fc2_b, nullptr, nullptr, nullptr, nullptr, out, nullptr);
}
